// rdf_61770219651753
// MI455X (gfx1250) — compile-verified, matrix operand order checked by reading, not run on hardware
//
#include <hip/hip_runtime.h>
#include <math.h>

typedef __attribute__((ext_vector_type(16))) _Float16 v16h;
typedef __attribute__((ext_vector_type(8)))  float    v8f;

#define NBINS   64
#define NATOMS  512
#define NBATCH  4
#define CUTOFF  7.5f
#define CELL    20.0f
#define HALF_CELL 10.0f
#define CB2     64.0f            /* (cutoff + 0.5)^2 */
#define LOG2E   1.4426950408889634f

// Raw v_exp_f32 (no subnormal-range fixup). Arg is always <= 0 here; deep
// tail underflows to 0.0, which is exactly the gaussian/mask semantics.
__device__ __forceinline__ float fast_exp2(float x) {
#if __has_builtin(__builtin_amdgcn_exp2f)
    return __builtin_amdgcn_exp2f(x);
#else
    return exp2f(x);
#endif
}

// ---------------------------------------------------------------------------
// Kernel 1: zero the 64-bin f32 accumulator in workspace (graph-replay safe).
// ---------------------------------------------------------------------------
__global__ void rdf_zero_kernel(float* __restrict__ hist) {
    if (threadIdx.x < NBINS) hist[threadIdx.x] = 0.0f;
}

// ---------------------------------------------------------------------------
// Kernel 2: pair distances + gaussian smear + WMMA ones-reduction into bins.
// One wave per (b, i) row: 2048 waves = 256 blocks x 8 waves.
// ---------------------------------------------------------------------------
__global__ __launch_bounds__(256)
void rdf_pairs_kernel(const float* __restrict__ xyz, float* __restrict__ hist) {
    __shared__ float lhist[NBINS];
    const int tid  = threadIdx.x;
    if (tid < NBINS) lhist[tid] = 0.0f;
    __syncthreads();

    const int lane = tid & 31;
    const int wave = tid >> 5;
    const int row  = blockIdx.x * 8 + wave;      // 0..2047
    const int b    = row >> 9;                   // batch
    const int i    = row & (NATOMS - 1);         // center atom

    const float* __restrict__ base = xyz + (size_t)b * NATOMS * 3;
    const float xi = base[i * 3 + 0];
    const float yi = base[i * 3 + 1];
    const float zi = base[i * 3 + 2];

    // Gaussian smearing constants (g_off spacing = cutoff/(nbins-1))
    const float width  = CUTOFF / (float)(NBINS - 1);
    const float coeff2 = (-0.5f / (width * width)) * LOG2E;   // for exp2
    const float mybin  = (float)(lane & 15);                  // bin within group

    // A matrix = all ones (f16), layout irrelevant since uniform.
    v16h ones;
#pragma unroll
    for (int t = 0; t < 16; ++t) ones[t] = (_Float16)1.0f;

    // Four f32 accumulators, one per 16-bin group (bins g*16 .. g*16+15).
    v8f acc[4];
#pragma unroll
    for (int g = 0; g < 4; ++g) acc[g] = (v8f){};

    for (int jc = 0; jc < NATOMS; jc += 32) {
        const int j = jc + lane;
        // minimum-image displacement
        float dx = base[j * 3 + 0] - xi;
        float dy = base[j * 3 + 1] - yi;
        float dz = base[j * 3 + 2] - zi;
        dx += ((dx < -HALF_CELL) ? CELL : 0.0f) - ((dx >= HALF_CELL) ? CELL : 0.0f);
        dy += ((dy < -HALF_CELL) ? CELL : 0.0f) - ((dy >= HALF_CELL) ? CELL : 0.0f);
        dz += ((dz < -HALF_CELL) ? CELL : 0.0f) - ((dz >= HALF_CELL) ? CELL : 0.0f);
        const float dsq = dx * dx + dy * dy + dz * dz;
        const bool  m   = (dsq < CB2) && (dsq != 0.0f);
        // fold mask into r: huge r -> raw v_exp_f32 underflows to exactly 0
        const float r   = m ? sqrtf(dsq) : 1.0e3f;

        // Redistribute r: lane L needs pairs (L&16)+0..15 of this chunk
        float rr[16];
        const int srcBase = lane & 16;
#pragma unroll
        for (int t = 0; t < 16; ++t) rr[t] = __shfl(r, srcBase + t, 32);

        // B matrix (32 pairs x 16 bins, f16): lane = column(bin), halves = K(pair)
#pragma unroll
        for (int g = 0; g < 4; ++g) {
            const float off = (mybin + 16.0f * (float)g) * width;
            v16h bm;
#pragma unroll
            for (int t = 0; t < 16; ++t) {
                const float x = rr[t] - off;
                bm[t] = (_Float16)fast_exp2(coeff2 * x * x);
            }
            // D = ones(16x32) * G(32x16) + C  -> per-bin sums, rows identical
            acc[g] = __builtin_amdgcn_wmma_f32_16x16x32_f16(
                false, ones, false, bm, (short)0, acc[g], false, false);
        }
    }

    // D layout: lane n (0..15), VGPR 0 = row M=0, col N=n  -> bin partial sum.
    if (lane < 16) {
#pragma unroll
        for (int g = 0; g < 4; ++g)
            atomicAdd(&lhist[g * 16 + (lane & 15)], acc[g][0]);
    }
    __syncthreads();
    if (tid < NBINS) atomicAdd(&hist[tid], lhist[tid]);
}

// ---------------------------------------------------------------------------
// Kernel 3: normalize + emit (count[64], bins[65], rdf[64]) = 193 floats.
// ---------------------------------------------------------------------------
__global__ void rdf_finalize_kernel(const float* __restrict__ hist,
                                    float* __restrict__ out) {
    const int t = threadIdx.x;
    float total = 0.0f;
#pragma unroll
    for (int k = 0; k < NBINS; ++k) total += hist[k];
    const float inv   = 1.0f / total;
    const float edgew = CUTOFF / (float)NBINS;   // bin-edge spacing
    if (t < NBINS) {
        const float c  = hist[t] * inv;
        out[t] = c;                              // count
        const float e0 = (float)t * edgew;
        const float e1 = (float)(t + 1) * edgew;
        const float denom = e1 * e1 * e1 - e0 * e0 * e0;
        // rdf = count * V / vol_bin = count * cutoff^3 / (e1^3 - e0^3)
        out[129 + t] = c * (CUTOFF * CUTOFF * CUTOFF) / denom;
    }
    if (t < NBINS + 1) out[64 + t] = (float)t * edgew;   // bins
}

// ---------------------------------------------------------------------------
extern "C" void kernel_launch(void* const* d_in, const int* in_sizes, int n_in,
                              void* d_out, int out_size, void* d_ws, size_t ws_size,
                              hipStream_t stream) {
    (void)in_sizes; (void)n_in; (void)out_size; (void)ws_size;
    const float* xyz  = (const float*)d_in[0];
    float*       out  = (float*)d_out;
    float*       hist = (float*)d_ws;            // 64 f32 accumulator

    rdf_zero_kernel<<<1, 64, 0, stream>>>(hist);
    rdf_pairs_kernel<<<256, 256, 0, stream>>>(xyz, hist);
    rdf_finalize_kernel<<<1, 128, 0, stream>>>(hist, out);
}
